// RiemannianManifold_65687229825096
// MI455X (gfx1250) — matrix operand order repaired and checked by reading, not compiled, not run on hardware
//
#include <hip/hip_runtime.h>
#include <hip/hip_bf16.h>

// CDNA5 / gfx1250. wave32. WMMA f32_16x16x32_f16.
//
// Phase 1: per-point metric factors (4096 blocks, VALU).
// Phase 2: pairwise distances as one f16 GEMM (N=1024, M=1024, K=2176) per
//          batch using v_wmma_f32_16x16x32_f16, epilogue sqrt(clamp(0.5*acc)).

typedef _Float16 v16h __attribute__((ext_vector_type(16)));
typedef _Float16 v8h  __attribute__((ext_vector_type(8)));
typedef float    v8f  __attribute__((ext_vector_type(8)));

#define LDK 2176   // 2114 features padded to a multiple of 32 (and 128)

// ---------------------------------------------------------------------------
// Kernel 1: per-point factor / feature-row construction
// pid < 2048  -> query point (b*1024+n), writes A row
// pid >= 2048 -> key point   (b*1024+m), writes B row
// A row: [vec(G_q) 1024 | vec(q qT) 1024 | -2*G_q q 32 | -2*q 32 | a | 1 | 0pad]
// B row: [vec(k kT) 1024 | vec(G_k) 1024 |    k     32 | G_k k 32 | 1 | b | 0pad]
// dot(A_n, B_m) = qGq q - 2 (Gq q).k + kT Gq k + qT Gk q - 2 q.(Gk k) + kT Gk k
//              = delta^T (G_q + G_k) delta   ;   dist^2 = 0.5 * dot
// ---------------------------------------------------------------------------
__global__ __launch_bounds__(256) void rm_factor_kernel(
    const float* __restrict__ Q, const float* __restrict__ Kp,
    const float* __restrict__ W1, const float* __restrict__ b1,
    const float* __restrict__ W2, const float* __restrict__ b2,
    _Float16* __restrict__ Aq, _Float16* __restrict__ Bk)
{
    const int pid = blockIdx.x;        // 0..4095
    const bool isQ = pid < 2048;
    const int p = isQ ? pid : pid - 2048;
    const float* x = (isQ ? Q : Kp) + (size_t)p * 32;
    _Float16* outrow = (isQ ? Aq : Bk) + (size_t)p * LDK;
    const int t = threadIdx.x;

    __shared__ float sx[32];
    __shared__ float sh[64];
    __shared__ float sL[1024];
    __shared__ float sG[1024];
    __shared__ float sc[32];
    __shared__ float sa;

    if (t < 32) sx[t] = x[t];
    __syncthreads();

    // h = silu(x @ W1^T + b1), W1 is (64,32) row-major
    if (t < 64) {
        float acc = b1[t];
        #pragma unroll
        for (int d = 0; d < 32; ++d) acc += sx[d] * W1[t * 32 + d];
        sh[t] = acc / (1.0f + expf(-acc));
    }
    __syncthreads();

    // raw = clip(h @ W2^T + b2, -5, 5); build masked L (tril + softplus diag)
    for (int j = t; j < 1024; j += 256) {
        float acc = b2[j];
        #pragma unroll
        for (int i = 0; i < 64; ++i) acc += sh[i] * W2[j * 64 + i];
        acc = fminf(fmaxf(acc, -5.0f), 5.0f);
        const int d = j >> 5, e = j & 31;
        float Lv;
        if (e > d)       Lv = 0.0f;                       // strict upper: zero
        else if (e < d)  Lv = acc;                        // strict lower: raw
        else {                                            // diag: softplus(raw+1)+1e-4
            const float v = acc + 1.0f;                   // v in [-4, 6]
            Lv = log1pf(expf(v)) + 1e-4f;
        }
        sL[j] = Lv;
    }
    __syncthreads();

    // G = L L^T (upper of L already zeroed, so full 32-sum is exact)
    for (int j = t; j < 1024; j += 256) {
        const int d = j >> 5, e = j & 31;
        float g = 0.0f;
        #pragma unroll
        for (int jj = 0; jj < 32; ++jj) g += sL[d * 32 + jj] * sL[e * 32 + jj];
        sG[j] = g;
    }
    __syncthreads();

    // c = G x,  a = x^T c
    if (t < 32) {
        float acc = 0.0f;
        #pragma unroll
        for (int e = 0; e < 32; ++e) acc += sG[t * 32 + e] * sx[e];
        sc[t] = acc;
    }
    __syncthreads();
    if (t == 0) {
        float acc = 0.0f;
        #pragma unroll
        for (int d = 0; d < 32; ++d) acc += sx[d] * sc[d];
        sa = acc;
    }
    __syncthreads();

    // Emit f16 feature row (zero padding beyond 2114)
    for (int k = t; k < LDK; k += 256) {
        float v;
        if (isQ) {
            if (k < 1024)        v = sG[k];
            else if (k < 2048) { int kk = k - 1024; v = sx[kk >> 5] * sx[kk & 31]; }
            else if (k < 2080)   v = -2.0f * sc[k - 2048];
            else if (k < 2112)   v = -2.0f * sx[k - 2080];
            else if (k == 2112)  v = sa;
            else if (k == 2113)  v = 1.0f;
            else                 v = 0.0f;
        } else {
            if (k < 1024)      { v = sx[k >> 5] * sx[k & 31]; }
            else if (k < 2048)   v = sG[k - 1024];
            else if (k < 2080)   v = sx[k - 2048];
            else if (k < 2112)   v = sc[k - 2080];
            else if (k == 2112)  v = 1.0f;
            else if (k == 2113)  v = sa;
            else                 v = 0.0f;
        }
        outrow[k] = (_Float16)v;
    }
}

// ---------------------------------------------------------------------------
// Kernel 2: WMMA GEMM  C[b,n,m] = sqrt(clip(0.5 * A_b[n,:].B_b[m,:], 1e-6, 1e6))
// Both A and B stored row-major over K (B is "B^T" of the math GEMM, which is
// exactly the K-major layout the 32x16 B fragment wants).
// ---------------------------------------------------------------------------

// A fragment (16x32 f16): lane l holds row (l&15); halves 0-7 at K = 8*(l>=16)+i,
// halves 8-15 at K = 16 + 8*(l>=16) + (i-8)  -> two contiguous 8-half chunks.
__device__ __forceinline__ v16h load_a_frag(const _Float16* __restrict__ base, int lane)
{
    const _Float16* p = base + (size_t)(lane & 15) * LDK + ((lane >> 4) << 3);
    v8h lo = *(const v8h*)p;
    v8h hi = *(const v8h*)(p + 16);
    return __builtin_shufflevector(lo, hi, 0, 1, 2, 3, 4, 5, 6, 7,
                                   8, 9, 10, 11, 12, 13, 14, 15);
}

// B fragment (32x16 f16): lane l holds column (l&15); halves 0-7 at K = 8*(l>=16)+i,
// halves 8-15 at K = 16 + 8*(l>=16) + (i-8)  -> two contiguous 8-half chunks.
__device__ __forceinline__ v16h load_b_frag(const _Float16* __restrict__ base, int lane)
{
    const _Float16* p = base + (size_t)(lane & 15) * LDK + ((lane >> 4) << 3);
    v8h lo = *(const v8h*)p;
    v8h hi = *(const v8h*)(p + 16);
    return __builtin_shufflevector(lo, hi, 0, 1, 2, 3, 4, 5, 6, 7,
                                   8, 9, 10, 11, 12, 13, 14, 15);
}

__device__ __forceinline__ float rm_epilogue(float acc)
{
    float d2 = fminf(fmaxf(0.5f * acc, 1e-6f), 1e6f);
    return sqrtf(d2);
}

__global__ __launch_bounds__(256) void rm_gemm_kernel(
    const _Float16* __restrict__ A, const _Float16* __restrict__ Bm,
    float* __restrict__ out)
{
    const int bz   = blockIdx.z;                 // batch
    const int lane = threadIdx.x & 31;
    const int w    = threadIdx.x >> 5;           // 8 waves per block
    // block tile 128x64, wave tile 32x32 (4x2 wave grid)
    const int rowBase = blockIdx.x * 128 + (w >> 1) * 32;
    const int colBase = blockIdx.y * 64  + (w & 1) * 32;

    const _Float16* Ab = A  + (size_t)bz * 1024 * LDK;
    const _Float16* Bb = Bm + (size_t)bz * 1024 * LDK;

    v8f c00 = {}, c01 = {}, c10 = {}, c11 = {};

    for (int k0 = 0; k0 < LDK; k0 += 32) {
        v16h a0 = load_a_frag(Ab + (size_t)rowBase * LDK + k0, lane);
        v16h a1 = load_a_frag(Ab + (size_t)(rowBase + 16) * LDK + k0, lane);
        v16h b0 = load_b_frag(Bb + (size_t)colBase * LDK + k0, lane);
        v16h b1 = load_b_frag(Bb + (size_t)(colBase + 16) * LDK + k0, lane);

        c00 = __builtin_amdgcn_wmma_f32_16x16x32_f16(false, a0, false, b0,
                                                     (short)0, c00, false, false);
        c01 = __builtin_amdgcn_wmma_f32_16x16x32_f16(false, a0, false, b1,
                                                     (short)0, c01, false, false);
        c10 = __builtin_amdgcn_wmma_f32_16x16x32_f16(false, a1, false, b0,
                                                     (short)0, c10, false, false);
        c11 = __builtin_amdgcn_wmma_f32_16x16x32_f16(false, a1, false, b1,
                                                     (short)0, c11, false, false);
    }

    // C/D layout: VGPR v holds row (v + 8*(lane>=16)), column (lane&15)
    float* outb = out + (size_t)bz * 1024 * 1024;
    const int col0 = colBase + (lane & 15);
    const int row0 = rowBase + ((lane >> 4) << 3);
    #pragma unroll
    for (int v = 0; v < 8; ++v) {
        outb[(size_t)(row0 + v)      * 1024 + col0]      = rm_epilogue(c00[v]);
        outb[(size_t)(row0 + v)      * 1024 + col0 + 16] = rm_epilogue(c01[v]);
        outb[(size_t)(row0 + 16 + v) * 1024 + col0]      = rm_epilogue(c10[v]);
        outb[(size_t)(row0 + 16 + v) * 1024 + col0 + 16] = rm_epilogue(c11[v]);
    }
}

// ---------------------------------------------------------------------------
extern "C" void kernel_launch(void* const* d_in, const int* in_sizes, int n_in,
                              void* d_out, int out_size, void* d_ws, size_t ws_size,
                              hipStream_t stream)
{
    (void)in_sizes; (void)n_in; (void)out_size; (void)ws_size;
    const float* Q  = (const float*)d_in[0];   // (2,1024,32)
    const float* K  = (const float*)d_in[1];   // (2,1024,32)
    const float* W1 = (const float*)d_in[2];   // (64,32)
    const float* b1 = (const float*)d_in[3];   // (64,)
    const float* W2 = (const float*)d_in[4];   // (1024,64)
    const float* b2 = (const float*)d_in[5];   // (1024,)
    float* out = (float*)d_out;                // (2,1024,1024)

    // Workspace: A features then B features, f16, 2*1024 rows x 2176 each side
    // = 17,825,792 bytes total.
    _Float16* Aq = (_Float16*)d_ws;
    _Float16* Bk = Aq + (size_t)2 * 1024 * LDK;

    rm_factor_kernel<<<4096, 256, 0, stream>>>(Q, K, W1, b1, W2, b2, Aq, Bk);
    rm_gemm_kernel<<<dim3(8, 16, 2), 256, 0, stream>>>(Aq, Bk, out);
}
